// Predictor_1864015806798
// MI455X (gfx1250) — compile-verified, matrix operand order checked by reading, not run on hardware
//
#include <hip/hip_runtime.h>

typedef _Float16 v16h __attribute__((ext_vector_type(16)));
typedef float    v8f  __attribute__((ext_vector_type(8)));

#define NLOC   469          // 350 + 91 + 28
#define NCLS   80
#define NFLAT  37520        // NLOC * NCLS
#define TOPK_N 1000

__device__ __forceinline__ float sigmf(float x) { return 1.0f / (1.0f + expf(-x)); }

// ---------------------------------------------------------------------------
// 1) decode: packed sort keys for all 37520 (loc,class) scores + 469 boxes
// ---------------------------------------------------------------------------
__global__ void __launch_bounds__(1024) fcos_decode(
    const float* s4, const float* l4, const float* c4,
    const float* s5, const float* l5, const float* c5,
    const float* s6, const float* l6, const float* c6,
    unsigned long long* keys, float* bbox)
{
    int gid = blockIdx.x * 1024 + threadIdx.x;

    if (gid < NFLAT) {
        int p  = gid / NCLS;          // global location
        int cc = gid % NCLS;          // class column (channel cc+1)
        const float* sc; const float* ct; int hw, base;
        if (p < 350)      { sc = s4; ct = c4; hw = 350; base = 0;   }
        else if (p < 441) { sc = s5; ct = c5; hw = 91;  base = 350; }
        else              { sc = s6; ct = c6; hw = 28;  base = 441; }
        int i = p - base;
        float s = sqrtf(sigmf(sc[(cc + 1) * hw + i]) * sigmf(ct[i]));
        float m = (s >= 0.05f) ? s : -1.0f;
        unsigned b   = __float_as_uint(m);
        unsigned key = (b & 0x80000000u) ? ~b : (b | 0x80000000u);
        keys[gid] = ((unsigned long long)key << 32) | (unsigned)(~(unsigned)gid);
    }

    if (gid < NLOC) {
        int p = gid;
        const float* lc; int hw, base, w; float scale;
        if (p < 350)      { lc = l4; hw = 350; base = 0;   w = 25; scale = 0.0625f;   }
        else if (p < 441) { lc = l5; hw = 91;  base = 350; w = 13; scale = 0.03125f;  }
        else              { lc = l6; hw = 28;  base = 441; w = 7;  scale = 0.015625f; }
        int i   = p - base;
        float x = ((float)(i % w) + 0.5f) / scale;
        float y = ((float)(i / w) + 0.5f) / scale;
        float l = lc[0 * hw + i] / scale, t = lc[1 * hw + i] / scale;
        float r = lc[2 * hw + i] / scale, bo = lc[3 * hw + i] / scale;
        bbox[p * 4 + 0] = x - l;  bbox[p * 4 + 1] = y - t;
        bbox[p * 4 + 2] = x + r;  bbox[p * 4 + 3] = y + bo;
    }
}

// ---------------------------------------------------------------------------
// 2) topk: 8-pass MSB radix-select (exact 1000th key) + bitonic sort of 1024
// ---------------------------------------------------------------------------
__global__ void __launch_bounds__(1024) fcos_topk(
    const unsigned long long* keys, float* topVal, unsigned* topIdx)
{
    __shared__ unsigned hist[256];
    __shared__ unsigned long long sh_prefix;
    __shared__ unsigned sh_rem, cnt;
    __shared__ unsigned long long buf[1024];
    int tid = threadIdx.x;
    if (tid == 0) { sh_prefix = 0ULL; sh_rem = TOPK_N; cnt = 0; }
    __syncthreads();

    for (int pass = 0; pass < 8; ++pass) {
        int shift = 56 - pass * 8;
        if (tid < 256) hist[tid] = 0;
        __syncthreads();
        unsigned long long pre = sh_prefix;
        for (int i = tid; i < NFLAT; i += 1024) {
            unsigned long long k = keys[i];
            bool act = (pass == 0) || (((k ^ pre) >> (shift + 8)) == 0);
            if (act) atomicAdd(&hist[(unsigned)((k >> shift) & 0xFF)], 1u);
        }
        __syncthreads();
        if (tid == 0) {
            unsigned rem = sh_rem;
            int b = 255;
            for (; b > 0; --b) { if (hist[b] >= rem) break; rem -= hist[b]; }
            sh_prefix = pre | ((unsigned long long)(unsigned)b << shift);
            sh_rem = rem;
        }
        __syncthreads();
    }

    unsigned long long kth = sh_prefix;
    for (int i = tid; i < NFLAT; i += 1024) {
        unsigned long long k = keys[i];
        if (k >= kth) {
            unsigned pos = atomicAdd(&cnt, 1u);
            if (pos < 1024) buf[pos] = k;
        }
    }
    __syncthreads();
    unsigned c = cnt; if (c > 1024u) c = 1024u;
    if ((unsigned)tid >= c) buf[tid] = 0ULL;
    __syncthreads();

    // bitonic sort, descending
    for (int k = 2; k <= 1024; k <<= 1) {
        for (int j = k >> 1; j > 0; j >>= 1) {
            int ixj = tid ^ j;
            if (ixj > tid) {
                unsigned long long a = buf[tid], b = buf[ixj];
                bool asc = ((tid & k) != 0);
                if (asc ? (a > b) : (a < b)) { buf[tid] = b; buf[ixj] = a; }
            }
            __syncthreads();
        }
    }

    unsigned long long v = buf[tid];
    unsigned key  = (unsigned)(v >> 32);
    unsigned idx  = ~(unsigned)(v & 0xFFFFFFFFULL);
    unsigned bits = (key & 0x80000000u) ? (key ^ 0x80000000u) : ~key;
    topVal[tid] = __uint_as_float(bits);
    topIdx[tid] = idx;
}

// ---------------------------------------------------------------------------
// 3) prep: gather boxes, max reduction, class-offset boxes, fp16 areas,
//    wave32-ballot validity words
// ---------------------------------------------------------------------------
__global__ void __launch_bounds__(1024) fcos_prep(
    const float* bbox, const float* topVal, const unsigned* topIdx,
    float4* offbox, _Float16* areaH, unsigned* validW)
{
    __shared__ float red[1024];
    int t = threadIdx.x;
    bool have = t < TOPK_N;
    float4 bx = make_float4(0.f, 0.f, 0.f, 0.f);
    int lab = 0;
    if (have) {
        unsigned idx = topIdx[t];
        unsigned p = idx / (unsigned)NCLS;
        lab = (int)(idx % (unsigned)NCLS);
        bx.x = bbox[p * 4 + 0]; bx.y = bbox[p * 4 + 1];
        bx.z = bbox[p * 4 + 2]; bx.w = bbox[p * 4 + 3];
    }
    red[t] = have ? fmaxf(fmaxf(bx.x, bx.y), fmaxf(bx.z, bx.w)) : -3.4e38f;
    __syncthreads();
    for (int s = 512; s > 0; s >>= 1) {
        if (t < s) red[t] = fmaxf(red[t], red[t + s]);
        __syncthreads();
    }
    float base = red[0] + 1.0f;
    float off  = (float)lab * base;
    offbox[t] = have ? make_float4(bx.x + off, bx.y + off, bx.z + off, bx.w + off)
                     : make_float4(0.f, 0.f, 0.f, 0.f);
    float ar = have ? fmaxf(bx.z - bx.x, 0.f) * fmaxf(bx.w - bx.y, 0.f) : 0.f;
    areaH[t] = (_Float16)ar;
    bool pred = have && (topVal[t] >= 0.05f);
    unsigned word = __builtin_amdgcn_ballot_w32(pred);
    if ((t & 31) == 0) validW[t >> 5] = word;
}

// ---------------------------------------------------------------------------
// 4) supmat: suppression bitmask. area_i + area_j outer sum via WMMA f16,
//    division-free IoU test per lane (inter > thresh*denom), ballot-packed.
//    grid = 64 blocks (16 rows each) x 1024 threads (wave w -> cols 32w..32w+31)
// ---------------------------------------------------------------------------
__global__ void __launch_bounds__(1024) fcos_supmat(
    const float4* offbox, const _Float16* areaH, unsigned* supmat)
{
    int tid  = threadIdx.x;
    int lane = tid & 31, wave = tid >> 5;
    int rowbase = blockIdx.x * 16;
    int c0 = wave * 32, c1 = wave * 32 + 16;

    v16h A, B0, B1;
#pragma unroll
    for (int q = 0; q < 16; ++q) { A[q] = (_Float16)0; B0[q] = (_Float16)0; B1[q] = (_Float16)0; }
    if (lane < 16) {
        // A row m: K0 = area_i, K1 = 1   (lanes 0-15 hold K=0..7,16..23)
        A[0]  = areaH[rowbase + lane];  A[1]  = (_Float16)1.0f;
        // B col n: K0 = 1, K1 = area_j  (lanes 0-15 hold K=0..7,16..23)
        B0[0] = (_Float16)1.0f;         B0[1] = areaH[c0 + lane];
        B1[0] = (_Float16)1.0f;         B1[1] = areaH[c1 + lane];
    }
    v8f C = {0.f, 0.f, 0.f, 0.f, 0.f, 0.f, 0.f, 0.f};
    v8f D0 = __builtin_amdgcn_wmma_f32_16x16x32_f16(false, A, false, B0, (short)0, C, false, false);
    v8f D1 = __builtin_amdgcn_wmma_f32_16x16x32_f16(false, A, false, B1, (short)0, C, false, false);

    int j0 = c0 + (lane & 15);
    int j1 = c1 + (lane & 15);
    float4 bj0 = offbox[j0];
    float4 bj1 = offbox[j1];

#pragma unroll
    for (int r = 0; r < 8; ++r) {
        int i = rowbase + r + ((lane < 16) ? 0 : 8);   // matches D VGPR layout
        float4 bi = offbox[i];

        // iou > 0.6  <=>  inter > 0.6 * max(sumArea - inter, 1e-9)  (denom > 0)
        float ix = fminf(bi.z, bj0.z) - fmaxf(bi.x, bj0.x);
        float iy = fminf(bi.w, bj0.w) - fmaxf(bi.y, bj0.y);
        float inter = fmaxf(ix, 0.f) * fmaxf(iy, 0.f);
        float den = fmaxf(D0[r] - inter, 1e-9f);
        bool p0 = (inter > 0.6f * den) && (j0 > i) && (j0 < TOPK_N) && (i < TOPK_N);
        unsigned w0 = __builtin_amdgcn_ballot_w32(p0);

        ix = fminf(bi.z, bj1.z) - fmaxf(bi.x, bj1.x);
        iy = fminf(bi.w, bj1.w) - fmaxf(bi.y, bj1.y);
        inter = fmaxf(ix, 0.f) * fmaxf(iy, 0.f);
        den = fmaxf(D1[r] - inter, 1e-9f);
        bool p1 = (inter > 0.6f * den) && (j1 > i) && (j1 < TOPK_N) && (i < TOPK_N);
        unsigned w1 = __builtin_amdgcn_ballot_w32(p1);

        if (lane == 0) {
            int ilo = rowbase + r, ihi = rowbase + r + 8;
            unsigned wlo = (w0 & 0xFFFFu) | ((w1 & 0xFFFFu) << 16);
            unsigned whi = (w0 >> 16)     | (w1 & 0xFFFF0000u);
            if (ilo < TOPK_N) supmat[ilo * 32 + wave] = wlo;
            if (ihi < TOPK_N) supmat[ihi * 32 + wave] = whi;
        }
    }
}

// ---------------------------------------------------------------------------
// 5) nms: async-stage bitmask into LDS (2 x 64KB phases), then a single
//    wave32 runs the exact sequential suppression loop with ds_bpermute.
// ---------------------------------------------------------------------------
__device__ __forceinline__ void async_copy_b32(unsigned lds_off, const unsigned* gptr)
{
    unsigned long long ga = (unsigned long long)(size_t)gptr;
    asm volatile("global_load_async_to_lds_b32 %0, %1, off"
                 :: "v"(lds_off), "v"(ga) : "memory");
}

__global__ void __launch_bounds__(1024) fcos_nms(
    const unsigned* supmat, const unsigned* validW, unsigned* keepW)
{
    __shared__ unsigned sup[16384];   // 64KB of the 320KB/WGP LDS
    int tid = threadIdx.x;
    unsigned kw = 0;
    if (tid < 32) kw = validW[tid];

    // phase A: rows 0..511  (words 0..16383)
    for (int w = tid; w < 16384; w += 1024)
        async_copy_b32((unsigned)(size_t)&sup[w], supmat + w);
    asm volatile("s_wait_asynccnt 0" ::: "memory");
    __syncthreads();

    if (tid < 32) {
        for (int i = 0; i < 512; ++i) {
            unsigned ki = (unsigned)__builtin_amdgcn_ds_bpermute((i >> 5) << 2, (int)kw);
            if ((ki >> (i & 31)) & 1u) kw &= ~sup[i * 32 + tid];
        }
    }
    __syncthreads();

    // phase B: rows 512..999 (words 16384..31999)
    for (int w = tid; w < 15616; w += 1024)
        async_copy_b32((unsigned)(size_t)&sup[w], supmat + 16384 + w);
    asm volatile("s_wait_asynccnt 0" ::: "memory");
    __syncthreads();

    if (tid < 32) {
        for (int i = 512; i < TOPK_N; ++i) {
            unsigned ki = (unsigned)__builtin_amdgcn_ds_bpermute((i >> 5) << 2, (int)kw);
            if ((ki >> (i & 31)) & 1u) kw &= ~sup[(i - 512) * 32 + tid];
        }
        keepW[tid] = kw;
    }
}

// ---------------------------------------------------------------------------
// 6) out: clip boxes, small-box filter, write (bx, sc, label, keep) flat
// ---------------------------------------------------------------------------
__global__ void __launch_bounds__(1024) fcos_out(
    const float* bbox, const float* topVal, const unsigned* topIdx,
    const unsigned* keepW, float* out)
{
    int t = threadIdx.x;
    if (t >= TOPK_N) return;
    unsigned idx = topIdx[t];
    unsigned p = idx / (unsigned)NCLS;
    int lab = (int)(idx % (unsigned)NCLS);
    float x1 = bbox[p * 4 + 0], y1 = bbox[p * 4 + 1];
    float x2 = bbox[p * 4 + 2], y2 = bbox[p * 4 + 3];
    float cx1 = fminf(fmaxf(x1, 0.f), 224.f);
    float cy1 = fminf(fmaxf(y1, 0.f), 398.f);
    float cx2 = fminf(fmaxf(x2, 0.f), 224.f);
    float cy2 = fminf(fmaxf(y2, 0.f), 398.f);
    bool k = ((keepW[t >> 5] >> (t & 31)) & 1u) != 0;
    k = k && ((cx2 - cx1) >= 1e-5f) && ((cy2 - cy1) >= 1e-5f);
    out[t * 4 + 0] = k ? cx1 : 0.f;
    out[t * 4 + 1] = k ? cy1 : 0.f;
    out[t * 4 + 2] = k ? cx2 : 0.f;
    out[t * 4 + 3] = k ? cy2 : 0.f;
    out[4000 + t] = k ? topVal[t] : -1.0f;
    out[5000 + t] = (float)lab;
    out[6000 + t] = k ? 1.0f : 0.0f;
}

// ---------------------------------------------------------------------------
extern "C" void kernel_launch(void* const* d_in, const int* in_sizes, int n_in,
                              void* d_out, int out_size, void* d_ws, size_t ws_size,
                              hipStream_t stream)
{
    (void)in_sizes; (void)n_in; (void)out_size; (void)ws_size;
    const float* s4 = (const float*)d_in[0];
    const float* l4 = (const float*)d_in[1];
    const float* c4 = (const float*)d_in[2];
    const float* s5 = (const float*)d_in[3];
    const float* l5 = (const float*)d_in[4];
    const float* c5 = (const float*)d_in[5];
    const float* s6 = (const float*)d_in[6];
    const float* l6 = (const float*)d_in[7];
    const float* c6 = (const float*)d_in[8];

    char* w = (char*)d_ws;
    unsigned long long* keys = (unsigned long long*)(w + 0);        // 300160 B
    float*     bbox   = (float*)    (w + 300160);                   //   7504 B
    float*     topVal = (float*)    (w + 307664);                   //   4096 B
    unsigned*  topIdx = (unsigned*) (w + 311760);                   //   4096 B
    float4*    offbox = (float4*)   (w + 315856);                   //  16384 B
    _Float16*  areaH  = (_Float16*) (w + 332240);                   //   2048 B
    unsigned*  validW = (unsigned*) (w + 334288);                   //    128 B
    unsigned*  keepW  = (unsigned*) (w + 334416);                   //    128 B
    unsigned*  supmat = (unsigned*) (w + 334544);                   // 128000 B

    float* out = (float*)d_out;

    fcos_decode<<<37, 1024, 0, stream>>>(s4, l4, c4, s5, l5, c5, s6, l6, c6, keys, bbox);
    fcos_topk  <<<1,  1024, 0, stream>>>(keys, topVal, topIdx);
    fcos_prep  <<<1,  1024, 0, stream>>>(bbox, topVal, topIdx, offbox, areaH, validW);
    fcos_supmat<<<64, 1024, 0, stream>>>(offbox, areaH, supmat);
    fcos_nms   <<<1,  1024, 0, stream>>>(supmat, validW, keepW);
    fcos_out   <<<1,  1024, 0, stream>>>(bbox, topVal, topIdx, keepW, out);
}
